// BinaryConv2d_20220706030268
// MI455X (gfx1250) — matrix operand order repaired and checked by reading, not compiled, not run on hardware
//
#include <hip/hip_runtime.h>
#include <stdint.h>

typedef __attribute__((ext_vector_type(16))) _Float16 v16h;
typedef __attribute__((ext_vector_type(8)))  _Float16 v8h;
typedef __attribute__((ext_vector_type(8)))  float    v8f;
typedef __attribute__((ext_vector_type(4)))  float    v4f;
typedef __attribute__((ext_vector_type(4)))  unsigned int v4u;
typedef __attribute__((ext_vector_type(8)))  int      v8i;
typedef __attribute__((ext_vector_type(4)))  int      v4i;

#define CIN    128
#define COUT   256
#define HW     112
#define KTOT   1152          // 3*3*128
#define KSTEP  32
#define MTILE  128
#define NTILE  128
#define MSPATIAL (32*112*112)  // 401408, divisible by MTILE
#define LDSPAD 40            // 32 halves + 8 pad -> 80B row stride, conflict-free

// --------------------------------------------------------------------------
// Kernel 1: binarize weights -> f16 {-1,+1}, layout Wq[o][k], k = kh*384+kw*128+c
// Source W is HWIO: idx = ((kh*3+kw)*128 + c)*256 + o, so k == idx/256 directly.
// --------------------------------------------------------------------------
__global__ __launch_bounds__(256) void quantize_weights(const float* __restrict__ W,
                                                        _Float16* __restrict__ Wq) {
    int idx = blockIdx.x * 256 + threadIdx.x;       // 0 .. 294911
    if (idx >= 9 * CIN * COUT) return;
    int o = idx % COUT;
    int k = idx / COUT;
    float w = W[idx];
    Wq[(size_t)o * KTOT + k] = (_Float16)((w >= 0.0f) ? 1.0f : -1.0f);
}

// --------------------------------------------------------------------------
// Kernel 2: implicit-GEMM binary conv via v_wmma_f32_16x16x32_f16.
// B-tile (binarized weights) is staged LDS-side by the Tensor Data Mover:
// 2D tile of 128 rows x 64B, row stride 2304B, TDM inserts 16B LDS padding
// after every 64B so the LDS row stride is 80B (bank-conflict-free).
// --------------------------------------------------------------------------
__global__ __launch_bounds__(256) void bconv_wmma(const float*    __restrict__ x,
                                                  const _Float16* __restrict__ Wq,
                                                  const float*    __restrict__ bias,
                                                  float*          __restrict__ out) {
    __shared__ _Float16 As[MTILE][LDSPAD];   // activations tile (f16), 10KB
    __shared__ _Float16 Bs[NTILE][LDSPAD];   // binarized weights tile (f16), 10KB

    const int tid    = threadIdx.x;
    const int mBlock = blockIdx.x * MTILE;
    const int nBlock = blockIdx.y * NTILE;

    // loader role for A: 2 threads per tile row, each handles 16 of 32 k-elems
    const int rowL  = tid >> 1;              // 0..127
    const int halfL = tid & 1;               // 0/1

    // decode spatial position of this thread's A row (fixed across k loop)
    const int m   = mBlock + rowL;
    const int nb  = m / (HW * HW);
    const int rem = m - nb * (HW * HW);
    const int h   = rem / HW;
    const int w   = rem - h * HW;

    // wave tiling: 4 M-waves x 2 N-waves; per wave: 2 M-sub x 4 N-sub 16x16 tiles
    const int wave  = tid >> 5;
    const int lane  = tid & 31;
    const int wm    = wave & 3;
    const int wn    = wave >> 2;
    const int laneM = lane & 15;
    const int laneG = lane >> 4;             // lane group 0/1

    v8f acc[2][4];
#pragma unroll
    for (int i = 0; i < 2; ++i)
#pragma unroll
        for (int j = 0; j < 4; ++j) acc[i][j] = {};

    // ---- TDM descriptor invariants (uniform) ----
    const uint32_t ldsB = (uint32_t)(uintptr_t)(&Bs[0][0]);   // low 32b of flat = LDS offset
    // group1: workgroup_mask=0, data_size=1 (2B), pad_enable, pad_interval=3
    // (pad after every 16 DWORDs = 64B), pad_amount=3 (4 DWORDs = 16B)
    const uint32_t g1w0 = (1u << 16) | (1u << 20) | (3u << 22) | (3u << 25);

    const int nSteps = KTOT / KSTEP;         // 36
    for (int ks = 0; ks < nSteps; ++ks) {
        const int k0  = ks * KSTEP;
        const int c0  = k0 & (CIN - 1);
        const int kw_ = (k0 / CIN) % 3;
        const int kh_ = k0 / (CIN * 3);
        const int ih  = h + kh_ - 1;
        const int iw  = w + kw_ - 1;

        // ---- load A chunk (16 floats -> 16 halves), zero for SAME padding ----
        v8h a0 = {}, a1 = {};
        if ((unsigned)ih < (unsigned)HW && (unsigned)iw < (unsigned)HW) {
            const float* src = x + ((((size_t)nb * HW + ih) * HW + iw) * CIN + c0 + halfL * 16);
            v4f f0 = *(const v4f*)(src + 0);
            v4f f1 = *(const v4f*)(src + 4);
            v4f f2 = *(const v4f*)(src + 8);
            v4f f3 = *(const v4f*)(src + 12);
            __builtin_prefetch(src + CIN, 0, 0);   // global_prefetch_b8 (next kw tap)
#pragma unroll
            for (int j = 0; j < 4; ++j) {
                a0[j]     = (_Float16)f0[j];
                a0[4 + j] = (_Float16)f1[j];
                a1[j]     = (_Float16)f2[j];
                a1[4 + j] = (_Float16)f3[j];
            }
        }

        __syncthreads();   // all waves done reading previous LDS tiles

        // ---- B tile via Tensor Data Mover (one DMA per block, wave 0 issues) ----
        if (wave == 0) {
            const uint64_t gaddr =
                (uint64_t)(uintptr_t)(Wq + (size_t)nBlock * KTOT + k0);
            v4u g0;
            g0[0] = 1u;                                        // count=1, user mode
            g0[1] = ldsB;                                      // lds_addr
            g0[2] = (uint32_t)gaddr;                           // global_addr[31:0]
            g0[3] = (uint32_t)((gaddr >> 32) & 0x1FFFFFFu)     // global_addr[56:32]
                    | (2u << 30);                              // type=2 ("image")
            v8i g1;
            g1[0] = (int)g1w0;
            g1[1] = (int)(((uint32_t)KTOT & 0xFFFFu) << 16);   // tensor_dim0 lo16
            g1[2] = (int)(((uint32_t)KTOT >> 16)               // tensor_dim0 hi16
                          | (((uint32_t)COUT & 0xFFFFu) << 16)); // tensor_dim1 lo16
            g1[3] = (int)(((uint32_t)COUT >> 16)               // tensor_dim1 hi16
                          | ((uint32_t)KSTEP << 16));          // tile_dim0 = 32
            g1[4] = NTILE;                                     // tile_dim1=128, tile_dim2=0
            g1[5] = KTOT;                                      // tensor_dim0_stride[31:0]
            g1[6] = 0;                                         // stride0 hi / stride1 lo
            g1[7] = 0;
            v4i g2 = {};
            v4i g3 = {};
            v8i g4 = {};
            __builtin_amdgcn_tensor_load_to_lds(g0, g1, g2, g3, g4, 0);
        }

        // ---- A tile into LDS (all 8 waves) ----
        *(v8h*)&As[rowL][halfL * 16 + 0] = a0;
        *(v8h*)&As[rowL][halfL * 16 + 8] = a1;

        if (wave == 0) {
            __builtin_amdgcn_s_wait_tensorcnt(0);   // B tile landed in LDS
        }
        __syncthreads();   // tiles visible to all waves

        // ---- build fragments and issue WMMAs ----
        v16h afrag[2];
#pragma unroll
        for (int i = 0; i < 2; ++i) {
            const int row = wm * 32 + i * 16 + laneM;
            // lane<16: K 0-7 and 16-23 ; lane>=16: K 8-15 and 24-31
            v8h lo = *(const v8h*)&As[row][laneG * 8];
            v8h hi = *(const v8h*)&As[row][laneG * 8 + 16];
#pragma unroll
            for (int j = 0; j < 8; ++j) { afrag[i][j] = lo[j]; afrag[i][8 + j] = hi[j]; }
        }
        v16h bfrag[4];
#pragma unroll
        for (int j2 = 0; j2 < 4; ++j2) {
            const int col = wn * 64 + j2 * 16 + laneM;
            // lane<16: K 0-7 and 16-23 ; lane>=16: K 8-15 and 24-31
            v8h lo = *(const v8h*)&Bs[col][laneG * 8];
            v8h hi = *(const v8h*)&Bs[col][laneG * 8 + 16];
#pragma unroll
            for (int j = 0; j < 8; ++j) { bfrag[j2][j] = lo[j]; bfrag[j2][8 + j] = hi[j]; }
        }
#pragma unroll
        for (int i = 0; i < 2; ++i)
#pragma unroll
            for (int j2 = 0; j2 < 4; ++j2)
                acc[i][j2] = __builtin_amdgcn_wmma_f32_16x16x32_f16(
                    /*neg_a=*/false, afrag[i],
                    /*neg_b=*/false, bfrag[j2],
                    /*c_mod=*/(short)0, acc[i][j2],
                    /*reuse_a=*/false, /*reuse_b=*/false);
    }

    // ---- epilogue: D layout -> global, add bias ----
#pragma unroll
    for (int j2 = 0; j2 < 4; ++j2) {
        const int n  = nBlock + wn * 64 + j2 * 16 + laneM;
        const float bv = bias[n];
#pragma unroll
        for (int i = 0; i < 2; ++i) {
            const int mbase = mBlock + wm * 32 + i * 16 + laneG * 8;
#pragma unroll
            for (int r = 0; r < 8; ++r) {
                out[(size_t)(mbase + r) * COUT + n] = acc[i][j2][r] + bv;
            }
        }
    }
}

// --------------------------------------------------------------------------
extern "C" void kernel_launch(void* const* d_in, const int* in_sizes, int n_in,
                              void* d_out, int out_size, void* d_ws, size_t ws_size,
                              hipStream_t stream) {
    const float* x = (const float*)d_in[0];
    const float* W = (const float*)d_in[1];
    const float* b = (const float*)d_in[2];
    float* out     = (float*)d_out;
    _Float16* Wq   = (_Float16*)d_ws;       // 256*1152 halves = 576 KB scratch

    // 1) binarize weights into GEMM-friendly [COUT][KTOT] f16 layout
    quantize_weights<<<(9 * CIN * COUT + 255) / 256, 256, 0, stream>>>(W, Wq);

    // 2) implicit-GEMM WMMA conv: grid = (M/128, N/128)
    dim3 grid(MSPATIAL / MTILE, COUT / NTILE);   // (3136, 2)
    bconv_wmma<<<grid, 256, 0, stream>>>(x, Wq, b, out);
}
